// ContradictionMonitor_54296976556624
// MI455X (gfx1250) — 8-bit matrix operand order repaired and checked by reading, compiled, not run on hardware
//
#include <hip/hip_runtime.h>
#include <math.h>

typedef __attribute__((ext_vector_type(8))) int v8i;

#define T_DIM 4096
#define D_DIM 1024
#define A_DIM 32
#define S_LEN 16
#define H_LEN 256
#define W_LEN 241
#define BA    256

__device__ inline int alias_rt(int t) {
  switch (t) {
    case 11: case 13: case 16: return 11;
    case 21: case 22: case 23: return 21;
    case 31: case 32: case 33: return 31;
    case 41: case 42: case 43: case 44: return 41;
    case 51: case 52: case 53: return 51;
    default: return -1;
  }
}
__device__ inline unsigned long long compat_mask(int r) {
  switch (r) {
    case 11: return (1ull<<11)|(1ull<<13)|(1ull<<16);
    case 21: return (1ull<<14)|(1ull<<15)|(1ull<<21)|(1ull<<22)|(1ull<<23);
    case 31: return (1ull<<15)|(1ull<<31)|(1ull<<32)|(1ull<<33);
    case 41: return (1ull<<41)|(1ull<<42)|(1ull<<43)|(1ull<<44);
    case 51: return (1ull<<15)|(1ull<<51)|(1ull<<52)|(1ull<<53);
    default: return 0ull;
  }
}
__device__ inline unsigned int onehot4(int tok, int kb) {
  unsigned int d = (unsigned int)(tok - kb);
  return (d < 4u) ? (1u << (8u * d)) : 0u;
}

__device__ inline float wave_allsum(float v) {
  #pragma unroll
  for (int m = 16; m > 0; m >>= 1) v += __shfl_xor(v, m, 32);
  return v;
}

__device__ inline float block_sum(float v, float* red) {
  const int t = threadIdx.x;
  red[t] = v; __syncthreads();
  for (int off = 128; off > 0; off >>= 1) {
    if (t < off) red[t] += red[t + off];
    __syncthreads();
  }
  float r = red[0]; __syncthreads();
  return r;
}
__device__ inline float block_max(float v, float* red) {
  const int t = threadIdx.x;
  red[t] = v; __syncthreads();
  for (int off = 128; off > 0; off >>= 1) {
    if (t < off) red[t] = fmaxf(red[t], red[t + off]);
    __syncthreads();
  }
  float r = red[0]; __syncthreads();
  return r;
}

__global__ __launch_bounds__(256) void hidden_kernel(
    const float* __restrict__ hidden, const float* __restrict__ anchor,
    const int* __restrict__ anchor_end, float* __restrict__ out,
    float* __restrict__ ws_hidden_c)
{
  __shared__ float sCol[D_DIM];
  __shared__ float sWdot[8];
  __shared__ float sWshift[8];
  __shared__ float sRed[256];

  const int idx  = blockIdx.x;
  const int b    = idx >> 5;
  const int t    = threadIdx.x;
  const int wid  = t >> 5;
  const int lane = t & 31;

  const int start = anchor_end[idx] + 1;
  const float* __restrict__ F  = hidden + ((size_t)b * T_DIM + (size_t)start) * D_DIM;
  const float* __restrict__ Av = anchor + (size_t)idx * D_DIM;

  #pragma unroll
  for (int k = 0; k < 4; ++k) sCol[t * 4 + k] = 0.0f;
  __syncthreads();

  float4 av[8];
  #pragma unroll
  for (int j = 0; j < 8; ++j)
    av[j] = *(const float4*)(Av + j * 128 + lane * 4);

  float na2p = 0.0f;
  #pragma unroll
  for (int j = 0; j < 8; ++j)
    na2p += av[j].x*av[j].x + av[j].y*av[j].y + av[j].z*av[j].z + av[j].w*av[j].w;
  const float na2 = wave_allsum(na2p);

  float4 csum[8];
  #pragma unroll
  for (int j = 0; j < 8; ++j) csum[j] = make_float4(0.f, 0.f, 0.f, 0.f);

  float dot_acc = 0.0f, shift_acc = 0.0f;

  for (int r = 0; r < 32; ++r) {
    const int h = wid + 8 * r;
    const float* __restrict__ row = F + (size_t)h * D_DIM;
    float dp = 0.0f, sp = 0.0f;
    #pragma unroll
    for (int j = 0; j < 8; ++j) {
      const float4 f = *(const float4*)(row + j * 128 + lane * 4);
      dp += av[j].x*f.x + av[j].y*f.y + av[j].z*f.z + av[j].w*f.w;
      sp += f.x*f.x + f.y*f.y + f.z*f.z + f.w*f.w;
      csum[j].x += f.x; csum[j].y += f.y; csum[j].z += f.z; csum[j].w += f.w;
    }
    const float dot_h = wave_allsum(dp);
    const float sq_h  = wave_allsum(sp);
    dot_acc   += dot_h;
    shift_acc += sqrtf(fmaxf(sq_h - 2.0f * dot_h + na2, 0.0f));
  }

  if (lane == 0) { sWdot[wid] = dot_acc; sWshift[wid] = shift_acc; }
  #pragma unroll
  for (int j = 0; j < 8; ++j) {
    const int c = j * 128 + lane * 4;
    atomicAdd(&sCol[c + 0], csum[j].x);
    atomicAdd(&sCol[c + 1], csum[j].y);
    atomicAdd(&sCol[c + 2], csum[j].z);
    atomicAdd(&sCol[c + 3], csum[j].w);
  }
  __syncthreads();

  float mp = 0.0f;
  #pragma unroll
  for (int k = 0; k < 4; ++k) {
    const float m = sCol[t * 4 + k] * (1.0f / 256.0f);
    mp += m * m;
  }
  const float mn2 = block_sum(mp, sRed);

  if (t == 0) {
    float dtot = 0.f, stot = 0.f;
    for (int w = 0; w < 8; ++w) { dtot += sWdot[w]; stot += sWshift[w]; }
    const float future_shift = stot * (1.0f / 256.0f);
    const float meandot      = dtot * (1.0f / 256.0f);
    const float nr = fmaxf(sqrtf(na2), 1e-8f);
    const float nf = fmaxf(sqrtf(mn2), 1e-8f);
    const float sim = meandot / (nr * nf);
    const float hidden_c = fmaxf(0.0f, 0.5f * (1.0f - sim));
    out[1 * BA + idx] = future_shift;
    out[2 * BA + idx] = sim;
    out[3 * BA + idx] = hidden_c;
    ws_hidden_c[idx]  = hidden_c;
  }
}

__global__ __launch_bounds__(256) void token_kernel(
    const int* __restrict__ input_ids, const int* __restrict__ anchor_end,
    float* __restrict__ out, const float* __restrict__ ws_hidden_c)
{
  __shared__ int   sTok[H_LEN];
  __shared__ int   sSpan[S_LEN];
  __shared__ int   sP[H_LEN][S_LEN];
  __shared__ int   sQ[H_LEN][4];
  __shared__ float sFirstF[S_LEN];
  __shared__ int   sRoot, sHasT;
  __shared__ float sDenom;
  __shared__ float sRed[256];

  const int idx = blockIdx.x;
  const int b   = idx >> 5;
  const int t   = threadIdx.x;
  const int ae  = anchor_end[idx];
  const int base = b * T_DIM;

  sTok[t] = input_ids[base + ae + 1 + t];
  if (t < S_LEN) sSpan[t] = input_ids[base + ae - (S_LEN - 1) + t];
  __syncthreads();

  if (t == 0) {
    int span[S_LEN];
    for (int s = 0; s < S_LEN; ++s) span[s] = sSpan[s];
    int first_root = -1;
    for (int s = 0; s < S_LEN; ++s) {
      const int al = alias_rt(span[s]);
      if (al >= 0) { first_root = al; break; }
    }
    int maxc = 0, counts[S_LEN];
    for (int s = 0; s < S_LEN; ++s) {
      int c = 0;
      for (int j = 0; j < S_LEN; ++j) c += (span[j] == span[s]) ? 1 : 0;
      counts[s] = c; if (c > maxc) maxc = c;
    }
    int mode = 64;
    for (int s = 0; s < S_LEN; ++s)
      if (counts[s] == maxc && span[s] < mode) mode = span[s];
    float denom = 0.0f;
    for (int s = 0; s < S_LEN; ++s) {
      bool first = true;
      for (int j = 0; j < s; ++j) if (span[j] == span[s]) { first = false; break; }
      sFirstF[s] = first ? 1.0f : 0.0f;
      denom += first ? 1.0f : 0.0f;
    }
    const int root = (first_root >= 0) ? first_root : mode;
    sRoot  = root;
    sHasT  = (root == 11 || root == 21 || root == 31 || root == 41 || root == 51) ? 1 : 0;
    sDenom = denom;
  }
  __syncthreads();

  const int root       = sRoot;
  const int anchor_tok = sSpan[S_LEN - 1];
  const int wv   = t >> 5;
  const int lane = t & 31;
  const int n    = lane & 15;
  const int half = lane >> 4;

  v8i b1, b2;
  #pragma unroll
  for (int j = 0; j < 8; ++j) {
    const int kb = ((j >> 1) * 16 + (j & 1) * 4) + half * 8;
    b1[j] = (int)onehot4(sSpan[n], kb);
    unsigned int v2 = 0u;
    if (n == 0) {
      v2 = onehot4(root, kb);
    } else if (n == 1) {
      #pragma unroll
      for (int i = 0; i < 4; ++i)
        v2 |= ((alias_rt(kb + i) == root) ? 1u : 0u) << (8 * i);
    } else if (n == 2) {
      const unsigned long long cm = compat_mask(root);
      #pragma unroll
      for (int i = 0; i < 4; ++i)
        v2 |= ((unsigned int)((cm >> (kb + i)) & 1ull)) << (8 * i);
    } else if (n == 3) {
      v2 = onehot4(anchor_tok, kb);
    }
    b2[j] = (int)v2;
  }

  const v8i cz = {0, 0, 0, 0, 0, 0, 0, 0};
  const int Abase[8] = {0, 4, 16, 20, 32, 36, 48, 52};

  #pragma unroll
  for (int ti = 0; ti < 2; ++ti) {
    const int ht  = wv + ti * 8;
    const int tok = sTok[ht * 16 + n];
    v8i afr;
    #pragma unroll
    for (int j = 0; j < 8; ++j)
      afr[j] = (int)onehot4(tok, Abase[j] + half * 8);

    v8i dP = __builtin_amdgcn_wmma_i32_16x16x64_iu8(false, afr, false, b1, cz, false, false);
    v8i dQ = __builtin_amdgcn_wmma_i32_16x16x64_iu8(false, afr, false, b2, cz, false, false);

    #pragma unroll
    for (int r = 0; r < 8; ++r) {
      const int row = ht * 16 + r + half * 8;
      sP[row][n] = dP[r];
      if (n < 4) sQ[row][n] = dQ[r];
    }
  }
  __syncthreads();

  const bool active = (t < W_LEN);
  float sims = 0.f, rp = 0.f, regime = 0.f;
  if (active) {
    const int w = t;
    float es = 0.f, ps = 0.f;
    #pragma unroll
    for (int s = 0; s < S_LEN; ++s) {
      const float p = (float)sP[w + s][s];
      es += p;
      ps += p * ((1.0f - 0.04f * (float)s) * (1.0f / 11.2f));
    }
    int rsum = 0, asum = 0, hsum = 0;
    #pragma unroll
    for (int j = 0; j < S_LEN; ++j) {
      rsum += sQ[w + j][0];
      asum += sQ[w + j][1];
      hsum += sQ[w + j][2];
    }
    float ov = 0.f;
    for (int s = 0; s < S_LEN; ++s) {
      int cnt = 0;
      #pragma unroll
      for (int j = 0; j < S_LEN; ++j) cnt += sP[w + j][s];
      ov += sFirstF[s] * (cnt > 0 ? 1.0f : 0.0f);
    }
    const float exact      = es * (1.0f / 16.0f);
    const float positional = ps;
    const float overlap    = ov / sDenom;
    rp = (float)rsum * (1.0f / 16.0f);
    const float ac = (float)asum * (1.0f / 16.0f);
    const float hd = (float)hsum * (1.0f / 16.0f);
    regime = sHasT ? (0.55f * hd + 0.2f * overlap + 0.15f * ac + 0.1f * rp)
                   : (0.45f * exact + 0.3f * overlap + 0.1f * ac + 0.15f * rp);
    sims = 0.25f * exact + 0.15f * overlap + 0.35f * positional + 0.25f * regime;
  }

  const float sum_sims = block_sum(active ? sims : 0.f, sRed);
  const float best     = block_max(active ? sims : -1e30f, sRed);
  const float sum_rp   = block_sum(active ? rp : 0.f, sRed);
  const float sum_rg   = block_sum(active ? regime : 0.f, sRed);
  const float cnt06    = block_sum((active && sims >= 0.6f) ? 1.f : 0.f, sRed);
  const float toksum   = block_sum((float)sQ[t][3], sRed);

  if (t == 0) {
    const float invW = 1.0f / (float)W_LEN;
    const float mean_sims = sum_sims * invW;
    const float mrp = sum_rp * invW;
    const float mrc = sum_rg * invW;
    const float dmass = cnt06 * invW;
    const float dcoh  = 0.6f * mean_sims + 0.25f * mrp + 0.15f * mrc;
    const float pattern_c = 1.0f - (0.6f * best + 0.2f * mrp + 0.2f * mrc);
    const float token_c   = 1.0f - toksum * (1.0f / 256.0f);
    const float hidden_c  = ws_hidden_c[idx];
    float contr = 0.2f * hidden_c + 0.2f * token_c + 0.6f * pattern_c;
    contr = fminf(fmaxf(contr, 0.0f), 1.0f);
    out[0 * BA + idx] = contr;
    out[4 * BA + idx] = token_c;
    out[5 * BA + idx] = pattern_c;
    out[6 * BA + idx] = dmass;
    out[7 * BA + idx] = dcoh;
  }
}

extern "C" void kernel_launch(void* const* d_in, const int* in_sizes, int n_in,
                              void* d_out, int out_size, void* d_ws, size_t ws_size,
                              hipStream_t stream) {
  (void)in_sizes; (void)n_in; (void)out_size; (void)ws_size;
  const float* hidden     = (const float*)d_in[0];
  const float* anchor     = (const float*)d_in[1];
  const int*   input_ids  = (const int*)d_in[2];
  const int*   anchor_end = (const int*)d_in[3];
  float* out = (float*)d_out;
  float* wsf = (float*)d_ws;

  hidden_kernel<<<BA, 256, 0, stream>>>(hidden, anchor, anchor_end, out, wsf);
  token_kernel<<<BA, 256, 0, stream>>>(input_ids, anchor_end, out, wsf);
}
